// RefinerModel_49649821942437
// MI455X (gfx1250) — matrix operand order repaired and checked by reading, not compiled, not run on hardware
//
#include <hip/hip_runtime.h>
#include <hip/hip_bf16.h>
#include <math.h>

// ---------------- problem constants ----------------
#define L_SEQ   20000
#define D_IN    2048
#define F_DIM   64
#define NUM_ACT 48
#define NFRAMES 32
#define NSAMP   3
#define NBLOCKS 5
#define HID     64
#define MAXS    48   // S <= 48 (sorted values in [0,48))

// ---------------- workspace layout ----------------
// ints (first 1KB):
#define META_S      0
#define META_RANK   16    // 48 ints : value -> segment id
#define META_START  64    // 48 ints
#define META_LEN    112   // 48 ints
#define META_PRED   160   // 48 ints
#define META_COUNT  208   // 48 ints (histogram)
// floats (offsets in floats from ws base; meta occupies first 256 floats worth):
#define KEYS_OFF    256
#define VALS_OFF    (KEYS_OFF + F_DIM * L_SEQ)
#define SCOR_OFF    (VALS_OFF + F_DIM * L_SEQ)
#define SEGF_OFF    (SCOR_OFF + L_SEQ)
#define HLIN_OFF    (SEGF_OFF + MAXS * F_DIM)
#define HLFE_OFF    (HLIN_OFF + MAXS * 2 * F_DIM)
#define RIN_OFF     (HLFE_OFF + F_DIM)
#define HOUT_OFF    (RIN_OFF + MAXS * 3 * F_DIM)
// end = HOUT_OFF + 2*MAXS*HID  (~10.3 MB total)

typedef __attribute__((ext_vector_type(16))) _Float16 v16h;
typedef __attribute__((ext_vector_type(8)))  float    v8f;

__device__ __forceinline__ float leaky01(float x) { return x > 0.f ? x : 0.1f * x; }
__device__ __forceinline__ float sigm(float x)    { return 1.f / (1.f + expf(-x)); }

// ============================================================
// K1: histogram of action values (segments == runs of a sorted array)
// ============================================================
__global__ __launch_bounds__(1024) void rf_hist(const int* __restrict__ aidx,
                                                int* __restrict__ meta) {
    __shared__ int bins[NUM_ACT];
    int tid = threadIdx.x;
    if (tid < NUM_ACT) bins[tid] = 0;
    __syncthreads();
    for (int l = tid; l < L_SEQ; l += 1024) {
        int v = aidx[l];
        if (v >= 0 && v < NUM_ACT) atomicAdd(&bins[v], 1);
    }
    __syncthreads();
    if (tid < NUM_ACT) meta[META_COUNT + tid] = bins[tid];
}

// ============================================================
// K2: serial scan over 48 bins -> S, rank LUT, starts, lens, pred
// ============================================================
__global__ void rf_segscan(int* __restrict__ meta) {
    if (threadIdx.x != 0 || blockIdx.x != 0) return;
    int S = 0, start = 0;
    for (int v = 0; v < NUM_ACT; ++v) {
        int c = meta[META_COUNT + v];
        if (c > 0) {
            meta[META_RANK + v]  = S;
            meta[META_PRED + S]  = v;   // mean of a constant run == the value
            meta[META_LEN + S]   = c;
            meta[META_START + S] = start;
            start += c;
            ++S;
        } else {
            meta[META_RANK + v] = 0;
        }
    }
    meta[META_S] = S;
}

// ============================================================
// K3: big fused GEMM:  keys = Wk @ x^T, values = Wv @ x^T  (f16 WMMA, f32 acc)
//   x memory layout: batch_input[d * L_SEQ + l]  ==> exactly B[k=d][n=l]
//   out layout: keys[f * L_SEQ + l]
// M = 64 (4 wmma tiles), N = 128 per block (16 per wave), K-chunks of 32.
// ============================================================
__global__ __launch_bounds__(256) void rf_gemm_kv(const float* __restrict__ xin,
                                                  const float* __restrict__ Wk,
                                                  const float* __restrict__ Wv,
                                                  float* __restrict__ keys,
                                                  float* __restrict__ vals) {
    __shared__ _Float16 sAk[F_DIM][32];      // Wk chunk  [f][k]
    __shared__ _Float16 sAv[F_DIM][32];      // Wv chunk
    __shared__ _Float16 sBT[128][34];        // x chunk transposed [n][k], padded

    const int tid   = threadIdx.x;
    const int wave  = tid >> 5;
    const int lane  = tid & 31;
    const int Lbase = blockIdx.x * 128;

    v8f acck[4] = {};
    v8f accv[4] = {};

    for (int kb = 0; kb < D_IN; kb += 32) {
        // ---- stage weights (f32 -> f16), 2048 elems each, 8/thread ----
        for (int i = tid; i < F_DIM * 32; i += 256) {
            int f = i >> 5, k = i & 31;
            sAk[f][k] = (_Float16)Wk[f * D_IN + kb + k];
            sAv[f][k] = (_Float16)Wv[f * D_IN + kb + k];
        }
        // ---- stage x chunk: 32 d-rows x 128 l-cols, coalesced over l ----
        for (int i = tid; i < 32 * 128; i += 256) {
            int d = i >> 7, n = i & 127;
            int l = Lbase + n;
            if (l >= L_SEQ) l = L_SEQ - 1;           // clamp (stores are masked)
            sBT[n][d] = (_Float16)xin[(size_t)(kb + d) * L_SEQ + l];
            if (kb + 32 < D_IN)                       // hint next chunk
                __builtin_prefetch(&xin[(size_t)(kb + 32 + d) * L_SEQ + l], 0, 0);
        }
        __syncthreads();

        // ---- B operand (shared by all 4 M-tiles of this wave) ----
        // lane<16: n=lane,    K(h) = h<8 ? h   : h+8   (0..7, 16..23)
        // lane>=16: n=lane-16, K(h) = h<8 ? h+8 : h+16  (8..15, 24..31)
        v16h b;
        {
            int n    = lane & 15;
            int koff = (lane >> 4) << 3;
            const _Float16* src = &sBT[wave * 16 + n][koff];
            #pragma unroll
            for (int h = 0; h < 16; ++h) b[h] = src[h < 8 ? h : h + 8];
        }

        // ---- A operands + WMMA, per ISA 16-bit A 16x32 layout ----
        // lane<16 : M=lane,    K(h) = h<8 ? h     : h+8   (0..7, 16..23)
        // lane>=16: M=lane-16, K(h) = h<8 ? h+8   : h+16  (8..15, 24..31)
        {
            int m  = lane & 15;
            int k0 = (lane < 16) ? 0 : 8;
            #pragma unroll
            for (int mt = 0; mt < 4; ++mt) {
                int row = mt * 16 + m;
                v16h ak, av;
                #pragma unroll
                for (int h = 0; h < 8; ++h) {
                    ak[h]     = sAk[row][k0 + h];
                    av[h]     = sAv[row][k0 + h];
                    ak[8 + h] = sAk[row][k0 + 16 + h];
                    av[8 + h] = sAv[row][k0 + 16 + h];
                }
                acck[mt] = __builtin_amdgcn_wmma_f32_16x16x32_f16(
                    false, ak, false, b, (short)0, acck[mt], false, false);
                accv[mt] = __builtin_amdgcn_wmma_f32_16x16x32_f16(
                    false, av, false, b, (short)0, accv[mt], false, false);
            }
        }
        __syncthreads();
    }

    // ---- store: C/D layout: VGPR r -> M = r + 8*(lane>=16), N = lane&15 ----
    int l = Lbase + wave * 16 + (lane & 15);
    if (l < L_SEQ) {
        int fhalf = (lane >> 4) * 8;
        #pragma unroll
        for (int mt = 0; mt < 4; ++mt) {
            #pragma unroll
            for (int r = 0; r < 8; ++r) {
                int f = mt * 16 + fhalf + r;
                keys[(size_t)f * L_SEQ + l] = acck[mt][r];
                vals[(size_t)f * L_SEQ + l] = accv[mt][r];
            }
        }
    }
}

// ============================================================
// K4: scores[l] = <keys[:,l], query_emb[pred[seg(l)]]> / 8
// ============================================================
__global__ __launch_bounds__(256) void rf_scores(const float* __restrict__ keys,
                                                 const int* __restrict__ aidx,
                                                 const int* __restrict__ meta,
                                                 const float* __restrict__ qemb,
                                                 float* __restrict__ scores) {
    int l = blockIdx.x * 256 + threadIdx.x;
    if (l >= L_SEQ) return;
    int seg = meta[META_RANK + aidx[l]];
    int p   = meta[META_PRED + seg];
    const float* q = qemb + p * F_DIM;
    float s = 0.f;
    #pragma unroll 8
    for (int f = 0; f < F_DIM; ++f) s += keys[(size_t)f * L_SEQ + l] * q[f];
    scores[l] = s * 0.125f;   // 1/sqrt(64)
}

// ============================================================
// K5: per-segment softmax + weighted value sum -> seg_feat, hl_inp
// one block per segment
// ============================================================
__global__ __launch_bounds__(256) void rf_segfeat(const float* __restrict__ scores,
                                                  const float* __restrict__ vals,
                                                  const int* __restrict__ meta,
                                                  const float* __restrict__ lemb,
                                                  float* __restrict__ seg_feat,
                                                  float* __restrict__ hl_inp) {
    int s = blockIdx.x;
    int S = meta[META_S];
    if (s >= S) return;
    int st = meta[META_START + s];
    int ln = meta[META_LEN + s];
    __shared__ float red[256];
    int tid = threadIdx.x;

    // 1) max
    float m = -3.0e38f;
    for (int i = tid; i < ln; i += 256) m = fmaxf(m, scores[st + i]);
    red[tid] = m; __syncthreads();
    for (int o = 128; o > 0; o >>= 1) {
        if (tid < o) red[tid] = fmaxf(red[tid], red[tid + o]);
        __syncthreads();
    }
    m = red[0]; __syncthreads();

    // 2) Z
    float z = 0.f;
    for (int i = tid; i < ln; i += 256) z += expf(scores[st + i] - m);
    red[tid] = z; __syncthreads();
    for (int o = 128; o > 0; o >>= 1) {
        if (tid < o) red[tid] += red[tid + o];
        __syncthreads();
    }
    z = red[0]; __syncthreads();

    // 3) weighted values: thread = (f, g) with f = tid/4, g = tid%4
    int f = tid >> 2, g = tid & 3;
    float acc = 0.f;
    for (int i = g; i < ln; i += 4)
        acc += vals[(size_t)f * L_SEQ + st + i] * expf(scores[st + i] - m);
    red[tid] = acc; __syncthreads();
    if (tid < F_DIM) {
        float t = (red[tid * 4] + red[tid * 4 + 1] + red[tid * 4 + 2] + red[tid * 4 + 3]) / z;
        seg_feat[s * F_DIM + tid] = t;
        hl_inp[s * 2 * F_DIM + tid] = t;
        int p = meta[META_PRED + s];
        hl_inp[s * 2 * F_DIM + F_DIM + tid] = lemb[p * F_DIM + tid];
    }
}

// ============================================================
// K6: SparseSampleEmbedder: sample 32 frames x3, conv0 + 5 res-blocks
// + maxpool pyramid (32->1), mean over samples -> hl_feat[64].
// Deterministic sorted sampling stands in for the JAX PRNG draw.
// Single block, everything in LDS.
// ============================================================
__global__ __launch_bounds__(256) void rf_conv(const float* __restrict__ hl_inp,
                                               const int* __restrict__ meta,
                                               const float* __restrict__ c0w,
                                               const float* __restrict__ c0b,
                                               const float* __restrict__ bw1,
                                               const float* __restrict__ bb1,
                                               const float* __restrict__ bw2,
                                               const float* __restrict__ bb2,
                                               float* __restrict__ hl_feat) {
    __shared__ float s_in[2 * F_DIM * NFRAMES];   // 128 x 32
    __shared__ float s_a[F_DIM * NFRAMES];        // 64 x 32
    __shared__ float s_b[F_DIM * NFRAMES];
    __shared__ float s_t[F_DIM * NFRAMES];
    __shared__ float s_facc[F_DIM];

    int tid = threadIdx.x;
    int S   = meta[META_S];
    if (tid < F_DIM) s_facc[tid] = 0.f;
    __syncthreads();

    for (int smp = 0; smp < NSAMP; ++smp) {
        // sorted deterministic sample of NFRAMES indices in [0, S)
        for (int i = tid; i < 2 * F_DIM * NFRAMES; i += 256) {
            int c = i >> 5, j = i & 31;
            int idx = ((2 * j + smp) * S) / (2 * NFRAMES);
            if (idx > S - 1) idx = S - 1;
            s_in[c * NFRAMES + j] = hl_inp[idx * 2 * F_DIM + c];
        }
        __syncthreads();

        // conv0: 128 -> 64 channels, k=3, SAME
        for (int pos = tid; pos < F_DIM * NFRAMES; pos += 256) {
            int c = pos >> 5, t = pos & 31;
            float acc = c0b[c];
            for (int ci = 0; ci < 2 * F_DIM; ++ci) {
                const float* w = &c0w[(c * 2 * F_DIM + ci) * 3];
                const float* in = &s_in[ci * NFRAMES];
                if (t > 0)           acc += w[0] * in[t - 1];
                                     acc += w[1] * in[t];
                if (t < NFRAMES - 1) acc += w[2] * in[t + 1];
            }
            s_a[pos] = acc;
        }
        __syncthreads();

        float* cur = s_a;
        float* alt = s_b;
        int wcur = NFRAMES;
        for (int blk = 0; blk < NBLOCKS; ++blk) {
            const float* w1 = bw1 + blk * F_DIM * F_DIM * 3;
            const float* b1 = bb1 + blk * F_DIM;
            const float* w2 = bw2 + blk * F_DIM * F_DIM * 3;
            const float* b2 = bb2 + blk * F_DIM;

            // t = leaky(conv1(cur))
            for (int pos = tid; pos < F_DIM * wcur; pos += 256) {
                int c = pos / wcur, t = pos % wcur;
                float acc = b1[c];
                for (int ci = 0; ci < F_DIM; ++ci) {
                    const float* w = &w1[(c * F_DIM + ci) * 3];
                    const float* in = &cur[ci * NFRAMES];
                    if (t > 0)        acc += w[0] * in[t - 1];
                                      acc += w[1] * in[t];
                    if (t < wcur - 1) acc += w[2] * in[t + 1];
                }
                s_t[c * NFRAMES + t] = leaky01(acc);
            }
            __syncthreads();

            // cur = leaky(cur + conv2(t))   (in place; conv2 reads only s_t)
            for (int pos = tid; pos < F_DIM * wcur; pos += 256) {
                int c = pos / wcur, t = pos % wcur;
                float acc = b2[c];
                for (int ci = 0; ci < F_DIM; ++ci) {
                    const float* w = &w2[(c * F_DIM + ci) * 3];
                    const float* in = &s_t[ci * NFRAMES];
                    if (t > 0)        acc += w[0] * in[t - 1];
                                      acc += w[1] * in[t];
                    if (t < wcur - 1) acc += w[2] * in[t + 1];
                }
                cur[c * NFRAMES + t] = leaky01(cur[c * NFRAMES + t] + acc);
            }
            __syncthreads();

            // maxpool stride 2 into alt
            int wh = wcur >> 1;
            for (int pos = tid; pos < F_DIM * wh; pos += 256) {
                int c = pos / wh, t = pos % wh;
                alt[c * NFRAMES + t] =
                    fmaxf(cur[c * NFRAMES + 2 * t], cur[c * NFRAMES + 2 * t + 1]);
            }
            __syncthreads();
            { float* tmp = cur; cur = alt; alt = tmp; }
            wcur = wh;
        }
        // wcur == 1
        if (tid < F_DIM) s_facc[tid] += cur[tid * NFRAMES];
        __syncthreads();
    }
    if (tid < F_DIM) hl_feat[tid] = s_facc[tid] * (1.f / (float)NSAMP);
}

// ============================================================
// K7: refine_in[s] = [seg_feat | label | hl_feat]  (S x 192)
// ============================================================
__global__ __launch_bounds__(192) void rf_build_rin(const float* __restrict__ hl_inp,
                                                    const float* __restrict__ hl_feat,
                                                    const int* __restrict__ meta,
                                                    float* __restrict__ rin) {
    int s = blockIdx.x;
    if (s >= meta[META_S]) return;
    int j = threadIdx.x;
    rin[s * 3 * F_DIM + j] =
        (j < 2 * F_DIM) ? hl_inp[s * 2 * F_DIM + j] : hl_feat[j - 2 * F_DIM];
}

// ============================================================
// K8: bidirectional GRU (torch gate order r,z,n). block 0 = fwd, 1 = bwd.
// ============================================================
__global__ __launch_bounds__(192) void rf_gru(const float* __restrict__ rin,
                                              const int* __restrict__ meta,
                                              const float* __restrict__ wih,
                                              const float* __restrict__ whh,
                                              const float* __restrict__ bih,
                                              const float* __restrict__ bhh,
                                              float* __restrict__ hout) {
    int dir = blockIdx.x;
    int S   = meta[META_S];
    const float* Wih = wih + dir * 3 * HID * 3 * F_DIM;   // [192][192]
    const float* Whh = whh + dir * 3 * HID * HID;         // [192][64]
    const float* Bih = bih + dir * 3 * HID;
    const float* Bhh = bhh + dir * 3 * HID;

    __shared__ float h[HID];
    __shared__ float x[3 * F_DIM];
    __shared__ float gi[3 * HID];
    __shared__ float gh[3 * HID];
    int tid = threadIdx.x;
    if (tid < HID) h[tid] = 0.f;
    __syncthreads();

    for (int t = 0; t < S; ++t) {
        int s = dir ? (S - 1 - t) : t;
        x[tid] = rin[s * 3 * F_DIM + tid];
        __syncthreads();
        float a = Bih[tid];
        for (int j = 0; j < 3 * F_DIM; ++j) a += Wih[tid * 3 * F_DIM + j] * x[j];
        float b = Bhh[tid];
        for (int j = 0; j < HID; ++j) b += Whh[tid * HID + j] * h[j];
        gi[tid] = a; gh[tid] = b;
        __syncthreads();
        if (tid < HID) {
            float r  = sigm(gi[tid] + gh[tid]);
            float zt = sigm(gi[HID + tid] + gh[HID + tid]);
            float n  = tanhf(gi[2 * HID + tid] + r * gh[2 * HID + tid]);
            float hn = (1.f - zt) * n + zt * h[tid];
            h[tid] = hn;
            hout[(dir * MAXS + s) * HID + tid] = hn;
        }
        __syncthreads();
    }
}

// ============================================================
// K9: refine_pred[s][a] = out_b[a] + <[h_f(s)|h_b(s)], out_w[a]>
// written to the front of d_out (flat [1,S,48])
// ============================================================
__global__ __launch_bounds__(NUM_ACT) void rf_outproj(const float* __restrict__ hout,
                                                      const int* __restrict__ meta,
                                                      const float* __restrict__ outw,
                                                      const float* __restrict__ outb,
                                                      float* __restrict__ out) {
    int s = blockIdx.x;
    if (s >= meta[META_S]) return;
    int a = threadIdx.x;
    const float* hf = &hout[s * HID];
    const float* hb = &hout[(MAXS + s) * HID];
    const float* w  = &outw[a * 2 * HID];
    float acc = outb[a];
    for (int j = 0; j < HID; ++j) acc += hf[j] * w[j];
    for (int j = 0; j < HID; ++j) acc += hb[j] * w[HID + j];
    out[s * NUM_ACT + a] = acc;
}

// ============================================================
// K10: rollout[a][l] = refine_pred[seg(l)][a], written at d_out + S*48
// ============================================================
__global__ __launch_bounds__(256) void rf_rollout(float* __restrict__ out,
                                                  const int* __restrict__ meta,
                                                  const int* __restrict__ aidx) {
    int idx = blockIdx.x * 256 + threadIdx.x;
    if (idx >= NUM_ACT * L_SEQ) return;
    int a = idx / L_SEQ;
    int l = idx - a * L_SEQ;
    int S = meta[META_S];
    int seg = meta[META_RANK + aidx[l]];
    out[S * NUM_ACT + (size_t)a * L_SEQ + l] = out[seg * NUM_ACT + a];
}

// ============================================================
// launch
// ============================================================
extern "C" void kernel_launch(void* const* d_in, const int* in_sizes, int n_in,
                              void* d_out, int out_size, void* d_ws, size_t ws_size,
                              hipStream_t stream) {
    (void)in_sizes; (void)n_in; (void)out_size; (void)ws_size;

    const int*   aidx = (const int*)  d_in[0];
    const float* xin  = (const float*)d_in[1];   // [2048][20000]
    const float* Wk   = (const float*)d_in[2];
    const float* Wv   = (const float*)d_in[3];
    const float* qemb = (const float*)d_in[4];
    const float* lemb = (const float*)d_in[5];
    const float* c0w  = (const float*)d_in[6];
    const float* c0b  = (const float*)d_in[7];
    const float* bw1  = (const float*)d_in[8];
    const float* bb1  = (const float*)d_in[9];
    const float* bw2  = (const float*)d_in[10];
    const float* bb2  = (const float*)d_in[11];
    const float* gwih = (const float*)d_in[12];
    const float* gwhh = (const float*)d_in[13];
    const float* gbih = (const float*)d_in[14];
    const float* gbhh = (const float*)d_in[15];
    const float* outw = (const float*)d_in[16];
    const float* outb = (const float*)d_in[17];

    float* out  = (float*)d_out;
    int*   meta = (int*)  d_ws;
    float* wsf  = (float*)d_ws;
    float* keys = wsf + KEYS_OFF;
    float* vals = wsf + VALS_OFF;
    float* scor = wsf + SCOR_OFF;
    float* segf = wsf + SEGF_OFF;
    float* hlin = wsf + HLIN_OFF;
    float* hlfe = wsf + HLFE_OFF;
    float* rin  = wsf + RIN_OFF;
    float* hout = wsf + HOUT_OFF;

    rf_hist   <<<1, 1024, 0, stream>>>(aidx, meta);
    rf_segscan<<<1, 1, 0, stream>>>(meta);

    const int nlb = (L_SEQ + 127) / 128;   // 157
    rf_gemm_kv<<<nlb, 256, 0, stream>>>(xin, Wk, Wv, keys, vals);

    rf_scores <<<(L_SEQ + 255) / 256, 256, 0, stream>>>(keys, aidx, meta, qemb, scor);
    rf_segfeat<<<MAXS, 256, 0, stream>>>(scor, vals, meta, lemb, segf, hlin);
    rf_conv   <<<1, 256, 0, stream>>>(hlin, meta, c0w, c0b, bw1, bb1, bw2, bb2, hlfe);
    rf_build_rin<<<MAXS, 192, 0, stream>>>(hlin, hlfe, meta, rin);
    rf_gru    <<<2, 192, 0, stream>>>(rin, meta, gwih, gwhh, gbih, gbhh, hout);
    rf_outproj<<<MAXS, NUM_ACT, 0, stream>>>(hout, meta, outw, outb, out);
    rf_rollout<<<(NUM_ACT * L_SEQ + 255) / 256, 256, 0, stream>>>(out, meta, aidx);
}
